// VariationalGate_65481071396783
// MI455X (gfx1250) — matrix operand order repaired and checked by reading, not compiled, not run on hardware
//
#include <hip/hip_runtime.h>
#include <hip/hip_bf16.h>

typedef __attribute__((ext_vector_type(16))) _Float16     v16h;
typedef __attribute__((ext_vector_type(8)))  float        v8f;
typedef __attribute__((ext_vector_type(8)))  unsigned int v8u;

#define N_WIRES 12
#define NSTATES 4096   // 2^N_WIRES
#define BATCH   8192

// One wave32 computes a 16(batch-rows) x 16(wires, 12 live) tile of
//   out = P @ S,  P[row][b] = cos^2+sin^2 of phase(row,b),
//   phase(row,b) = sum_w phi[row][w] * bit(b,w)
// accumulated over K=4096 basis states in chunks of 32 via
// V_WMMA_F32_16X16X32_F16 (128 WMMAs per tile).
//
// Key factorization: b = kb | K (kb multiple of 32, K<32, disjoint bits)
//   phase  = phHigh(kb) + phLow(K-slot)          (phLow loop-invariant)
//   S sign = baseSign(K) ^ chunkParity(kb)       (base fragment loop-invariant,
//                                                 flip = XOR 0x8000 per half)
__global__ __launch_bounds__(128)
void vgate_wmma_kernel(const float* __restrict__ x,
                       const float* __restrict__ params,
                       float* __restrict__ out)
{
    const int lane = threadIdx.x & 31;
    const int wave = threadIdx.x >> 5;
    const int tile = blockIdx.x * 4 + wave;          // 16-row batch tile
    const int rowBase = tile * 16;

    const int rl = lane & 15;    // A: row within tile / B: column (wire)
    const int hi = lane >> 4;    // lane-half selector for K striping

    // phi[w] = DEPTH * x[row][w] + params[w] + params[w + N_WIRES]
    const int row = rowBase + rl;
    float phi[N_WIRES];
#pragma unroll
    for (int w = 0; w < N_WIRES; ++w)
        phi[w] = 2.0f * x[row * N_WIRES + w] + params[w] + params[w + N_WIRES];

    // ---- Loop-invariant A low-phase per slot s=(g*4+p)*2+e,
    //      K(s) = g*16 + hi*8 + 2p + e ; bit j of b -> wire 11-j (j=0..4 here)
    float phLow[16];
#pragma unroll
    for (int g = 0; g < 2; ++g)
#pragma unroll
        for (int p = 0; p < 4; ++p)
#pragma unroll
            for (int e = 0; e < 2; ++e) {
                const int K = g * 16 + hi * 8 + p * 2 + e;
                float v = 0.0f;
#pragma unroll
                for (int j = 0; j < 5; ++j)
                    v += (K & (1 << j)) ? phi[11 - j] : 0.0f;
                phLow[(g * 4 + p) * 2 + e] = v;
            }

    // ---- Loop-invariant B base fragment (packed f16 pairs as u32 words).
    // col = rl; S[b][col] = (-1)^popc(b >> (11-col)). Dead cols (rl>=12) -> ±0.
    const unsigned shift = (unsigned)(N_WIRES - 1 - rl) & 31u;
    const unsigned live  = (rl < N_WIRES) ? 0x3C00u : 0u;   // f16 1.0 or 0
    v8u bbase;
#pragma unroll
    for (int v = 0; v < 8; ++v) {
        const unsigned K0 = (unsigned)((v >> 2) * 16 + hi * 8 + 2 * (v & 3));
        const unsigned p0 = __popc(K0 >> shift) & 1u;
        const unsigned p1 = __popc((K0 + 1u) >> shift) & 1u;
        const unsigned wlo = live | (p0 << 15);
        const unsigned whi = live | (p1 << 15);
        bbase[v] = wlo | (whi << 16);
    }

    v8f acc = {};

    for (int kb = 0; kb < NSTATES; kb += 32) {
        // phHigh: bits 5..11 of kb -> wires 6..0 (uniform selects)
        float phHigh = 0.0f;
#pragma unroll
        for (int j = 5; j < N_WIRES; ++j)
            phHigh += (kb & (1 << j)) ? phi[11 - j] : 0.0f;

        // A fragment: honest cos^2+sin^2 per slot
        v16h afrag;
#pragma unroll
        for (int s = 0; s < 16; ++s) {
            const float ph = phHigh + phLow[s];
            float sn, cs;
            __sincosf(ph, &sn, &cs);
            afrag[s] = (_Float16)(cs * cs + sn * sn);
        }

        // B fragment: base signs flipped by chunk parity (sign-bit XOR)
        const unsigned pkb  = __popc((unsigned)kb >> shift) & 1u;
        const unsigned flip = pkb ? 0x80008000u : 0u;
        v8u bw;
#pragma unroll
        for (int v = 0; v < 8; ++v) bw[v] = bbase[v] ^ flip;
        const v16h bfrag = __builtin_bit_cast(v16h, bw);

        // 8 args: (neg_a, A, neg_b, B, c_mod, C, reuse_a, reuse_b)
        acc = __builtin_amdgcn_wmma_f32_16x16x32_f16(
            false, afrag, false, bfrag, (short)0, acc, false, false);
    }

    // ---- Store D: VGPR v -> row = v + hi*8 (within tile), col = rl
    const float scale = 1.0f / (float)NSTATES;
    if (rl < N_WIRES) {
#pragma unroll
        for (int v = 0; v < 8; ++v) {
            const int r = rowBase + v + hi * 8;
            out[r * N_WIRES + rl] = acc[v] * scale;
        }
    }
}

extern "C" void kernel_launch(void* const* d_in, const int* in_sizes, int n_in,
                              void* d_out, int out_size, void* d_ws, size_t ws_size,
                              hipStream_t stream) {
    const float* x      = (const float*)d_in[0];   // [8192, 12] f32
    const float* params = (const float*)d_in[1];   // [24] f32
    float*       out    = (float*)d_out;           // [8192, 12] f32
    (void)in_sizes; (void)n_in; (void)out_size; (void)d_ws; (void)ws_size;

    // 512 batch tiles of 16 rows; 4 waves (128 threads) per block
    const int tiles  = BATCH / 16;
    const int blocks = tiles / 4;
    vgate_wmma_kernel<<<blocks, 128, 0, stream>>>(x, params, out);
}
